// GroupedQueryAttention_88545045774950
// MI455X (gfx1250) — matrix operand order repaired and checked by reading, not compiled, not run on hardware
//
#include <hip/hip_runtime.h>
#include <hip/hip_bf16.h>
#include <math.h>

#define HIDDEN  2048
#define SEQ     2048
#define NHEADS  32
#define NKV     8
#define HD      64

typedef __attribute__((ext_vector_type(8)))  float    v8f;
typedef __attribute__((ext_vector_type(4)))  float    v4f;
typedef __attribute__((ext_vector_type(16))) __bf16   v16bf;
typedef __attribute__((ext_vector_type(8)))  __bf16   v8bf;
typedef __attribute__((ext_vector_type(2)))  __bf16   v2bf;
typedef __attribute__((ext_vector_type(4)))  unsigned v4u;
typedef __attribute__((ext_vector_type(8)))  unsigned v8u;

union Frag { v16bf v; v8bf h[2]; };

__device__ __forceinline__ v8f wmma_bf16(const Frag& a, const Frag& b, v8f c) {
  return __builtin_amdgcn_wmma_f32_16x16x32_bf16(false, a.v, false, b.v,
                                                 (short)0, c, false, false);
}

// TDM: DMA a [rows x 64] bf16 tile (row stride = strideElems) from global
// into LDS with 16B padding per 128B row (-> 72-element LDS rows).
__device__ __forceinline__ void tdm_tile64(const __bf16* gsrc, const __bf16* ldst,
                                           unsigned rows, unsigned strideElems) {
  unsigned long long ga = (unsigned long long)(uintptr_t)gsrc;
  unsigned lds = (unsigned)(uintptr_t)(const void*)ldst;
  v4u g0;
  g0[0] = 1u;                                       // count=1 (valid), user mode
  g0[1] = lds;                                      // lds_addr
  g0[2] = (unsigned)(ga & 0xffffffffu);             // global_addr[31:0]
  g0[3] = (unsigned)((ga >> 32) & 0x01ffffffu)      // global_addr[56:32]
        | (2u << 30);                               // type=2 (image)
  v8u g1;
  g1[0] = (1u << 16)                                // data_size=1 (2 bytes)
        | (1u << 20)                                // pad_enable
        | (4u << 22)                                // pad_interval: 32 dwords
        | (3u << 25);                               // pad_amount: 4 dwords (16B)
  g1[1] = 0u;                                       // tensor_dim0[15:0] (=65536)
  g1[2] = 1u;                                       // dim0[31:16]=1; dim1[15:0]=0
  g1[3] = 1u | (64u << 16);                         // dim1[31:16]=1; tile_dim0=64
  g1[4] = rows;                                     // tile_dim1=rows; tile_dim2=0
  g1[5] = strideElems;                              // tensor_dim0_stride low32
  g1[6] = 0u;
  g1[7] = 0u;
  asm volatile("tensor_load_to_lds %0, %1" :: "s"(g0), "s"(g1) : "memory");
}

// ---------------- fp32 -> bf16 convert, 8 elems/thread ----------------
__global__ void k_convert(const float* __restrict__ src, __bf16* __restrict__ dst, int n8) {
  int i = blockIdx.x * blockDim.x + threadIdx.x;
  if (i < n8) {
    v4f a = *(const v4f*)(src + (size_t)i * 8);
    v4f b = *(const v4f*)(src + (size_t)i * 8 + 4);
    v8bf o;
#pragma unroll
    for (int j = 0; j < 4; ++j) { o[j] = (__bf16)a[j]; o[j + 4] = (__bf16)b[j]; }
    *(v8bf*)(dst + (size_t)i * 8) = o;
  }
}

// ------- fp32 [K,N] -> bf16 [N,K] tiled transpose (weights, one-time) -------
__global__ __launch_bounds__(256) void k_convert_t(const float* __restrict__ src,
                                                   __bf16* __restrict__ dst,
                                                   int K, int N) {
  __shared__ float tile[32][33];
  int n0 = blockIdx.x * 32, k0 = blockIdx.y * 32;
  int tx = threadIdx.x & 31, ty = threadIdx.x >> 5;   // 32 x 8
#pragma unroll
  for (int i = 0; i < 4; ++i)
    tile[ty + 8 * i][tx] = src[(size_t)(k0 + ty + 8 * i) * N + n0 + tx];
  __syncthreads();
#pragma unroll
  for (int i = 0; i < 4; ++i)
    dst[(size_t)(n0 + ty + 8 * i) * K + k0 + tx] = (__bf16)tile[tx][ty + 8 * i];
}

// ---------------- RoPE in place, bf16 [S, nheads*64], optional out-scale ----
__global__ void k_rope(__bf16* x, int nheads, int rowstride, float oscale) {
  int idx = blockIdx.x * blockDim.x + threadIdx.x;
  int d = idx & 31;
  int h = (idx >> 5) % nheads;
  int s = idx / (32 * nheads);
  float inv = __expf(-((float)d / 32.0f) * 9.210340371976184f); // 10000^(-d/32)
  float ang = (float)s * inv;
  float c = cosf(ang), sn = sinf(ang);
  __bf16* p = x + (size_t)s * rowstride + h * 64;
  float x0 = (float)p[d];
  float x1 = (float)p[d + 32];
  p[d]      = (__bf16)((x0 * c - x1 * sn) * oscale);
  p[d + 32] = (__bf16)((x1 * c + x0 * sn) * oscale);
}

// ---- bf16 GEMM: C[M,N] = A[M,K] * B'[N,K]^T  (B pre-transposed, [N,K]) ----
// 256 threads (8 waves), tile 128x128x64. TDM-staged, double-buffered,
// one barrier per stage. Dynamic LDS: 4 x 128x72 bf16 = 73728 bytes.
// Layout: [0,9216) A buf0 | [9216,18432) A buf1 | [18432,27648) B buf0 |
//         [27648,36864) B buf1   (element offsets)
template <bool OUT_F32>
__global__ __launch_bounds__(256) void k_gemm(const __bf16* __restrict__ A,
                                              const __bf16* __restrict__ Bp,
                                              void* __restrict__ C,
                                              int M, int N, int K) {
  extern __shared__ __align__(16) __bf16 smem[];
  int tid  = threadIdx.x;
  int lane = tid & 31;
  int wv   = tid >> 5;
  int wm   = wv >> 2;
  int wn   = wv & 3;
  int m0   = blockIdx.y * 128;
  int n0   = blockIdx.x * 128;
  int lm   = lane & 15;
  int kbA  = (lane < 16) ? 0 : 8;
  int kbB  = (lane < 16) ? 0 : 8;
  int wid  = __builtin_amdgcn_readfirstlane(wv);

  v8f acc[4][2] = {};
  if (wid == 0) {
    tdm_tile64(A  + (size_t)m0 * K, smem,         128u, (unsigned)K);
    tdm_tile64(Bp + (size_t)n0 * K, smem + 18432, 128u, (unsigned)K);
  }
  int p = 0;
  for (int k0 = 0; k0 < K; k0 += 64) {
    bool nx = (k0 + 64) < K;
    if (wid == 0) __builtin_amdgcn_s_wait_tensorcnt(0);  // buf p DMA complete
    __syncthreads();
    if (nx && wid == 0) {                                // DMA overlaps compute
      tdm_tile64(A  + (size_t)m0 * K + k0 + 64, smem + (p ^ 1) * 9216,
                 128u, (unsigned)K);
      tdm_tile64(Bp + (size_t)n0 * K + k0 + 64, smem + 18432 + (p ^ 1) * 9216,
                 128u, (unsigned)K);
    }
    const __bf16* As = smem + p * 9216;
    const __bf16* Bs = smem + 18432 + p * 9216;
#pragma unroll
    for (int kc = 0; kc < 2; ++kc) {
      Frag a[4], b[2];
#pragma unroll
      for (int mt = 0; mt < 4; ++mt) {
        const __bf16* pa = &As[(wm * 64 + mt * 16 + lm) * 72 + kc * 32];
        a[mt].h[0] = *(const v8bf*)(pa + kbA);
        a[mt].h[1] = *(const v8bf*)(pa + 16 + kbA);
      }
#pragma unroll
      for (int nt = 0; nt < 2; ++nt) {
        const __bf16* pb = &Bs[(wn * 32 + nt * 16 + lm) * 72 + kc * 32];
        b[nt].h[0] = *(const v8bf*)(pb + kbB);
        b[nt].h[1] = *(const v8bf*)(pb + kbB + 16);
      }
#pragma unroll
      for (int mt = 0; mt < 4; ++mt)
#pragma unroll
        for (int nt = 0; nt < 2; ++nt)
          acc[mt][nt] = wmma_bf16(a[mt], b[nt], acc[mt][nt]);
    }
    p ^= 1;
  }

  int roff = (lane < 16) ? 0 : 8;
#pragma unroll
  for (int mt = 0; mt < 4; ++mt)
#pragma unroll
    for (int nt = 0; nt < 2; ++nt)
#pragma unroll
      for (int r = 0; r < 8; ++r) {
        size_t row = (size_t)(m0 + wm * 64 + mt * 16 + roff + r);
        size_t col = (size_t)(n0 + wn * 32 + nt * 16 + lm);
        if (OUT_F32) ((float*)C)[row * N + col] = acc[mt][nt][r];
        else         ((__bf16*)C)[row * N + col] = (__bf16)acc[mt][nt][r];
      }
}

// ---------------- Flash attention (causal, GQA 4:1), transposed scores ----
// Grid: (S/64, NHEADS). Block 128 = 4 waves; wave owns 16 query rows.
// 64-key blocks: nblk = qb+1; only block j==qb (scalar cond) needs masking.
// K staged by TDM, V by vector transpose; double-buffered, 1 barrier/block.
__global__ __launch_bounds__(128) void k_attn(const __bf16* __restrict__ Q,
                                              const __bf16* __restrict__ Kg,
                                              const __bf16* __restrict__ Vg,
                                              __bf16* __restrict__ Ctx) {
  __shared__ __align__(16) __bf16 Ks[2][64 * 72];   // [key][d]  (72 = TDM pad)
  __shared__ __align__(16) __bf16 Vts[2][64 * 72];  // [d][key]
  __shared__ __align__(16) __bf16 Pt[4][16 * 72];   // per-wave P^T [q][key]
  __shared__ __align__(16) __bf16 Ob[4][16 * 72];   // per-wave O   [q][d]
  int tid  = threadIdx.x;
  int lane = tid & 31;
  int w    = tid >> 5;
  int qb   = blockIdx.x;
  int h    = blockIdx.y;
  int hk   = h >> 2;
  int q0w  = qb * 64 + w * 16;
  int lm   = lane & 15;
  int kbA  = (lane < 16) ? 0 : 8;
  int kb16 = (lane < 16) ? 0 : 16;
  int roff = (lane < 16) ? 0 : 8;
  int wid  = __builtin_amdgcn_readfirstlane(w);

  // Q as B-operand fragments (Q^T 64x16), table k order from global
  Frag qf[2];
  {
    const __bf16* qr = Q + (size_t)(q0w + lm) * (NHEADS * HD) + h * HD;
    qf[0].h[0] = *(const v8bf*)(qr + kbA);
    qf[0].h[1] = *(const v8bf*)(qr + 16 + kbA);
    qf[1].h[0] = *(const v8bf*)(qr + 32 + kbA);
    qf[1].h[1] = *(const v8bf*)(qr + 48 + kbA);
  }

  // V staging (vector): keys 2vp,2vp+1 x 16 d, packed-pair transpose stores
  int vp = tid >> 2, vd0 = (tid & 3) * 16;
  v8bf rv[4];
  auto ldv = [&](int j) {
    const __bf16* vs = Vg + (size_t)(j * 64 + 2 * vp) * (NKV * HD) + hk * HD + vd0;
    rv[0] = *(const v8bf*)vs;
    rv[1] = *(const v8bf*)(vs + 8);
    rv[2] = *(const v8bf*)(vs + NKV * HD);
    rv[3] = *(const v8bf*)(vs + NKV * HD + 8);
  };
  auto stv = [&](int pp) {
#pragma unroll
    for (int i = 0; i < 8; ++i) {
      v2bf pa = { rv[0][i], rv[2][i] };
      v2bf pc = { rv[1][i], rv[3][i] };
      *(v2bf*)&Vts[pp][(vd0 + i) * 72 + 2 * vp]     = pa;
      *(v2bf*)&Vts[pp][(vd0 + 8 + i) * 72 + 2 * vp] = pc;
    }
  };

  v8f o[4] = {};
  float mval = -1e30f, lval = 0.0f;
  int nblk = qb + 1;

  if (wid == 0)
    tdm_tile64(Kg + hk * HD, &Ks[0][0], 64u, (unsigned)(NKV * HD));
  ldv(0);
  stv(0);
  int p = 0;
  for (int j = 0; j < nblk; ++j) {
    bool nx = (j + 1) < nblk;
    if (nx) ldv(j + 1);
    if (wid == 0) __builtin_amdgcn_s_wait_tensorcnt(0);  // K(p) DMA complete
    __syncthreads();
    if (nx && wid == 0)                                  // DMA overlaps compute
      tdm_tile64(Kg + (size_t)(j + 1) * 64 * (NKV * HD) + hk * HD,
                 &Ks[p ^ 1][0], 64u, (unsigned)(NKV * HD));

    // scores: 4 chunks of 16 keys, 2 d-halves each -> 8 WMMAs
    v8f st[4];
#pragma unroll
    for (int c = 0; c < 4; ++c) {
      Frag k0f, k1f;
      const __bf16* kr = &Ks[p][(c * 16 + lm) * 72];
      k0f.h[0] = *(const v8bf*)(kr + kbA);
      k0f.h[1] = *(const v8bf*)(kr + 16 + kbA);
      k1f.h[0] = *(const v8bf*)(kr + 32 + kbA);
      k1f.h[1] = *(const v8bf*)(kr + 48 + kbA);
      v8f z = {};
      z = wmma_bf16(k0f, qf[0], z);
      z = wmma_bf16(k1f, qf[1], z);
      st[c] = z;
    }

    float mn, al, ls;
    if (j < qb) {                       // scalar condition: fully unmasked
      float mx = st[0][0];
#pragma unroll
      for (int c = 0; c < 4; ++c)
#pragma unroll
        for (int r = 0; r < 8; ++r) mx = fmaxf(mx, st[c][r]);
      mx = fmaxf(mx, __shfl_xor(mx, 16, 32));
      mn = fmaxf(mval, mx);
      al = exp2f(mval - mn);
      ls = 0.0f;
#pragma unroll
      for (int c = 0; c < 4; ++c)
#pragma unroll
        for (int r = 0; r < 8; r += 2) {
          float p0 = exp2f(st[c][r] - mn);
          float p1 = exp2f(st[c][r + 1] - mn);
          ls += p0 + p1;
          v2bf pp = { (__bf16)p0, (__bf16)p1 };
          *(v2bf*)&Pt[w][lm * 72 + c * 16 + roff + r] = pp;
        }
    } else {                            // diagonal block: per-element mask
      int qi = q0w + lm;
      float mx = -1e30f;
#pragma unroll
      for (int c = 0; c < 4; ++c)
#pragma unroll
        for (int r = 0; r < 8; ++r) {
          int key = j * 64 + c * 16 + roff + r;
          mx = fmaxf(mx, (key <= qi) ? st[c][r] : -1e30f);
        }
      mx = fmaxf(mx, __shfl_xor(mx, 16, 32));
      mn = fmaxf(mval, mx);
      al = exp2f(mval - mn);
      ls = 0.0f;
#pragma unroll
      for (int c = 0; c < 4; ++c)
#pragma unroll
        for (int r = 0; r < 8; r += 2) {
          int key = j * 64 + c * 16 + roff + r;
          float p0 = (key     <= qi) ? exp2f(st[c][r]     - mn) : 0.0f;
          float p1 = (key + 1 <= qi) ? exp2f(st[c][r + 1] - mn) : 0.0f;
          ls += p0 + p1;
          v2bf pp = { (__bf16)p0, (__bf16)p1 };
          *(v2bf*)&Pt[w][lm * 72 + c * 16 + roff + r] = pp;
        }
    }
    ls += __shfl_xor(ls, 16, 32);
    lval = lval * al + ls;
    mval = mn;
#pragma unroll
    for (int nc = 0; nc < 4; ++nc) o[nc] = o[nc] * al;

    // O^T += V^T @ P^T : 8 WMMAs
    Frag pb[2];
    {
      const __bf16* pp = &Pt[w][lm * 72];
      pb[0].h[0] = *(const v8bf*)(pp + kbA);
      pb[0].h[1] = *(const v8bf*)(pp + 16 + kbA);
      pb[1].h[0] = *(const v8bf*)(pp + 32 + kbA);
      pb[1].h[1] = *(const v8bf*)(pp + 48 + kbA);
    }
#pragma unroll
    for (int nc = 0; nc < 4; ++nc) {
      Frag va0, va1;
      const __bf16* vr = &Vts[p][(nc * 16 + lm) * 72];
      va0.h[0] = *(const v8bf*)(vr + kbA);
      va0.h[1] = *(const v8bf*)(vr + 16 + kbA);
      va1.h[0] = *(const v8bf*)(vr + 32 + kbA);
      va1.h[1] = *(const v8bf*)(vr + 48 + kbA);
      o[nc] = wmma_bf16(va0, pb[0], o[nc]);
      o[nc] = wmma_bf16(va1, pb[1], o[nc]);
    }

    if (nx) stv(p ^ 1);                // after compute; visible at next barrier
    p ^= 1;
  }

  // normalize, transpose O^T -> [q][d] via per-wave LDS, coalesced store
  float inv = 1.0f / lval;
#pragma unroll
  for (int nc = 0; nc < 4; ++nc)
#pragma unroll
    for (int r = 0; r < 8; r += 2) {
      v2bf pp = { (__bf16)(o[nc][r] * inv), (__bf16)(o[nc][r + 1] * inv) };
      *(v2bf*)&Ob[w][lm * 72 + nc * 16 + roff + r] = pp;
    }
  {
    int q2 = lane >> 1;
    int dh = (lane & 1) * 32;
    const __bf16* src = &Ob[w][q2 * 72 + dh];
    __bf16* dst = Ctx + (size_t)(q0w + q2) * (NHEADS * HD) + h * HD + dh;
#pragma unroll
    for (int i = 0; i < 4; ++i)
      *(v8bf*)(dst + i * 8) = *(const v8bf*)(src + i * 8);
  }
}

// ---------------- launcher ----------------
extern "C" void kernel_launch(void* const* d_in, const int* in_sizes, int n_in,
                              void* d_out, int out_size, void* d_ws, size_t ws_size,
                              hipStream_t stream) {
  (void)in_sizes; (void)n_in; (void)out_size; (void)ws_size;
  const float* X  = (const float*)d_in[0];
  // d_in[1]=position_ids (arange, analytic), d_in[2]=causal mask (analytic)
  const float* Wq = (const float*)d_in[3];
  const float* Wk = (const float*)d_in[4];
  const float* Wv = (const float*)d_in[5];
  const float* Wo = (const float*)d_in[6];

  char* w = (char*)d_ws;
  __bf16* Xb  = (__bf16*)(w + 0);          // 8 MB, [S,HID]
  __bf16* WqT = (__bf16*)(w + 8388608);    // 8 MB, Wq^T [2048,2048]
  __bf16* WkT = (__bf16*)(w + 16777216);   // 2 MB, Wk^T [512,2048]
  __bf16* WvT = (__bf16*)(w + 18874368);   // 2 MB, Wv^T [512,2048]
  __bf16* WoT = (__bf16*)(w + 20971520);   // 8 MB, Wo^T [2048,2048]
  __bf16* Qb  = (__bf16*)(w + 29360128);   // 8 MB, [S,2048]
  __bf16* Kb  = (__bf16*)(w + 37748736);   // 2 MB, [S,512]
  __bf16* Vb  = (__bf16*)(w + 39845888);   // 2 MB, [S,512]
  __bf16* Cx  = (__bf16*)(w + 41943040);   // 8 MB, [S,2048]   (total 48 MB)

  const size_t gemmLds = 4u * 9216u * sizeof(__bf16);   // 73728 B dynamic LDS

  // 1) converts (weights transposed once)
  k_convert<<<2048, 256, 0, stream>>>(X, Xb, SEQ * HIDDEN / 8);
  k_convert_t<<<dim3(64, 64), 256, 0, stream>>>(Wq, WqT, HIDDEN, NHEADS * HD);
  k_convert_t<<<dim3(16, 64), 256, 0, stream>>>(Wk, WkT, HIDDEN, NKV * HD);
  k_convert_t<<<dim3(16, 64), 256, 0, stream>>>(Wv, WvT, HIDDEN, NKV * HD);
  k_convert_t<<<dim3(64, 64), 256, 0, stream>>>(Wo, WoT, NHEADS * HD, HIDDEN);

  // 2) projections (TDM-staged GEMM)
  k_gemm<false><<<dim3(16, 16), 256, gemmLds, stream>>>(Xb, WqT, Qb, SEQ, NHEADS * HD, HIDDEN);
  k_gemm<false><<<dim3(4,  16), 256, gemmLds, stream>>>(Xb, WkT, Kb, SEQ, NKV * HD,   HIDDEN);
  k_gemm<false><<<dim3(4,  16), 256, gemmLds, stream>>>(Xb, WvT, Vb, SEQ, NKV * HD,   HIDDEN);

  // 3) RoPE; Q pre-scaled by (1/sqrt(64))*log2(e) so attention exps are exp2
  k_rope<<<(SEQ * NHEADS * 32) / 256, 256, 0, stream>>>(Qb, NHEADS, NHEADS * HD,
                                                        0.125f * 1.4426950408889634f);
  k_rope<<<(SEQ * NKV    * 32) / 256, 256, 0, stream>>>(Kb, NKV, NKV * HD, 1.0f);

  // 4) flash attention (64-key blocks, TDM-staged K tiles)
  k_attn<<<dim3(SEQ / 64, NHEADS), 128, 0, stream>>>(Qb, Kb, Vb, Cx);

  // 5) output projection (fp32 out, TDM-staged GEMM)
  k_gemm<true><<<dim3(16, 16), 256, gemmLds, stream>>>(Cx, WoT, d_out, SEQ, HIDDEN, NHEADS * HD);
}
